// SubjectGenerator_19267223290397
// MI455X (gfx1250) — compile-verified, matrix operand order checked by reading, not run on hardware
//
#include <hip/hip_runtime.h>

// CDNA5 (gfx1250) wave32 WMMA LSTM.
// word:[T,B,27] f32 -> per-wave 16-batch tile, T=32 recurrence fully in regs.
// Gates computed transposed: G^T[64,16] = Wpad[64,32] @ {x,h}^T[32,16] via
// v_wmma_f32_16x16x32_f16 (4 gate tiles x 2 matmuls per step).
// Batch lives on the WMMA N axis => C/D layout and B layout keep each batch
// element in the same lane; the h->B repack is per-lane f32->f16 only.
// Activations use v_exp_f32 + v_rcp_f32 (TRANS pipe) instead of IEEE divides.

typedef __attribute__((ext_vector_type(16))) _Float16 v16h;
typedef __attribute__((ext_vector_type(8)))  float    v8f;

#define TSTEPS 32
#define NIN    27
#define NH     10

__device__ __forceinline__ float sigmoid_f(float x) {
    // 1/(1+e^-x) with hardware v_rcp_f32 (~1 ULP): e -> [0, inf), rcp(inf)=0 ok.
    float e = __expf(-x);
    return __builtin_amdgcn_rcpf(1.0f + e);
}
__device__ __forceinline__ float tanh_f(float x) {
    // tanh(|x|) = (1-e)/(1+e), e = exp(-2|x|) in (0,1]; one exp + one rcp.
    float e = __expf(-2.0f * fabsf(x));
    float r = __builtin_amdgcn_rcpf(1.0f + e);
    return copysignf((1.0f - e) * r, x);
}

__global__ __launch_bounds__(256) void lstm_wmma_kernel(
    const float* __restrict__ word,  // [T, B, 27]
    const float* __restrict__ W_ih,  // [40, 27]
    const float* __restrict__ W_hh,  // [40, 10]
    const float* __restrict__ b_ih,  // [40]
    const float* __restrict__ b_hh,  // [40]
    const float* __restrict__ W_fc,  // [5, 10]
    const float* __restrict__ b_fc,  // [5]
    const float* __restrict__ W_out, // [1, 5]
    const float* __restrict__ b_out, // [1]
    float* __restrict__ out,         // [B]
    int Bsz)
{
    __shared__ float lds_h[8][16][NH];

    const int lane = threadIdx.x & 31;
    const int wave = threadIdx.x >> 5;
    const int half = lane >> 4;       // 0: lanes 0-15, 1: lanes 16-31
    const int m    = lane & 15;       // row-in-tile (A/C) or batch col (B/C)
    const int b0   = (blockIdx.x * 8 + wave) * 16;
    if (b0 >= Bsz) return;

    // ---- Hoist weights into WMMA A-matrix registers (f16, zero-padded). ----
    // A 16x32 f16 layout: lane m = M row; elems 0..7 -> K = half*8 + e,
    // elems 8..15 -> K = 16 + half*8 + (e-8).
    v16h A_ih[4], A_hh[4];
    v8f  cbias[4];
    const bool vrow = (m < NH);
#pragma unroll
    for (int g = 0; g < 4; ++g) {
        const int row = g * NH + m;   // padded gate tile: 16 rows, 10 valid
#pragma unroll
        for (int e = 0; e < 16; ++e) {
            const int k = (e < 8) ? (half * 8 + e) : (16 + half * 8 + (e - 8));
            A_ih[g][e] = (_Float16)((vrow && k < NIN) ? W_ih[row * NIN + k] : 0.0f);
            A_hh[g][e] = (_Float16)((vrow && k < NH)  ? W_hh[row * NH  + k] : 0.0f);
        }
        // C/D layout: VGPR r holds M = r + 8*half, broadcast over batch lanes.
#pragma unroll
        for (int r = 0; r < 8; ++r) {
            const int mm = r + 8 * half;
            cbias[g][r] = (mm < NH) ? (b_ih[g * NH + mm] + b_hh[g * NH + mm]) : 0.0f;
        }
    }

    // ---- Recurrent state: h,c in C/D layout (VGPR r -> M=r+8*half, lane -> batch).
    v8f h_ = {};
    v8f c_ = {};

    for (int t = 0; t < TSTEPS; ++t) {
        // B-matrix for x^T [32(pad from 27) x 16]: lane = batch, elems 0..7 ->
        // K = half*8 + e, elems 8..15 -> K = 16 + half*8 + (e-8). Streamed once:
        // use non-temporal loads.
        const float* p = word + ((size_t)t * (size_t)Bsz + (size_t)(b0 + m)) * NIN;
        v16h xB;
        if (half == 0) {
#pragma unroll
            for (int e = 0; e < 8; ++e)  xB[e]     = (_Float16)__builtin_nontemporal_load(p + e);        // feat 0..7
#pragma unroll
            for (int e = 0; e < 8; ++e)  xB[8 + e] = (_Float16)__builtin_nontemporal_load(p + 16 + e);   // feat 16..23
        } else {
#pragma unroll
            for (int e = 0; e < 8; ++e)  xB[e]     = (_Float16)__builtin_nontemporal_load(p + 8 + e);    // feat 8..15
            xB[8]  = (_Float16)__builtin_nontemporal_load(p + 24);
            xB[9]  = (_Float16)__builtin_nontemporal_load(p + 25);
            xB[10] = (_Float16)__builtin_nontemporal_load(p + 26);
#pragma unroll
            for (int e = 11; e < 16; ++e) xB[e] = (_Float16)0.0f;
        }

        // B-matrix for h^T: same-lane repack of the f32 state (K=0..15 valid,
        // upper half zero; padded rows of h are exactly 0 by construction).
        v16h hB;
#pragma unroll
        for (int e = 0; e < 8; ++e)  hB[e] = (_Float16)h_[e];
#pragma unroll
        for (int e = 8; e < 16; ++e) hB[e] = (_Float16)0.0f;

        // G^T tiles: acc = W_ih_pad @ x^T + (W_hh_pad @ h^T + bias)
        v8f acc[4];
#pragma unroll
        for (int g = 0; g < 4; ++g) {
            acc[g] = __builtin_amdgcn_wmma_f32_16x16x32_f16(
                false, A_hh[g], false, hB, (short)0, cbias[g], false, false);
            acc[g] = __builtin_amdgcn_wmma_f32_16x16x32_f16(
                false, A_ih[g], false, xB, (short)0, acc[g], false, false);
        }

        // Elementwise gate nonlinearities + state update (per-lane, no shuffles).
#pragma unroll
        for (int r = 0; r < 8; ++r) {
            float iv = sigmoid_f(acc[0][r]);
            float fv = sigmoid_f(acc[1][r]);
            float gv = tanh_f(acc[2][r]);
            float ov = sigmoid_f(acc[3][r]);
            float cn = fv * c_[r] + iv * gv;
            float hn = ov * tanh_f(cn);
            c_[r] = fmaxf(cn, 0.0f);   // reference ReLUs both carried states
            h_[r] = fmaxf(hn, 0.0f);
        }
    }

    // ---- FC head: gather h[batch][0..9] through LDS, 16 lanes finish. ----
#pragma unroll
    for (int r = 0; r < 8; ++r) {
        const int mm = r + 8 * half;
        if (mm < NH) lds_h[wave][m][mm] = h_[r];
    }
    __syncthreads();

    if (half == 0) {
        float hv[NH];
#pragma unroll
        for (int k = 0; k < NH; ++k) hv[k] = lds_h[wave][m][k];
        float z = b_out[0];
#pragma unroll
        for (int j = 0; j < 5; ++j) {
            float y = b_fc[j];
#pragma unroll
            for (int k = 0; k < NH; ++k) y = fmaf(W_fc[j * NH + k], hv[k], y);
            y = fmaxf(y, 0.0f);
            z = fmaf(W_out[j], y, z);
        }
        out[b0 + m] = sigmoid_f(z);
    }
}

extern "C" void kernel_launch(void* const* d_in, const int* in_sizes, int n_in,
                              void* d_out, int out_size, void* d_ws, size_t ws_size,
                              hipStream_t stream) {
    (void)n_in; (void)d_ws; (void)ws_size; (void)out_size;
    const float* word  = (const float*)d_in[0];
    const float* W_ih  = (const float*)d_in[1];
    const float* W_hh  = (const float*)d_in[2];
    const float* b_ih  = (const float*)d_in[3];
    const float* b_hh  = (const float*)d_in[4];
    const float* W_fc  = (const float*)d_in[5];
    const float* b_fc  = (const float*)d_in[6];
    const float* W_out = (const float*)d_in[7];
    const float* b_out = (const float*)d_in[8];
    float* out = (float*)d_out;

    const int Bsz = in_sizes[0] / (TSTEPS * NIN);     // 65536
    const int batch_per_block = 8 * 16;               // 8 waves x 16 batch
    const int blocks = (Bsz + batch_per_block - 1) / batch_per_block;

    lstm_wmma_kernel<<<blocks, 256, 0, stream>>>(
        word, W_ih, W_hh, b_ih, b_hh, W_fc, b_fc, W_out, b_out, out, Bsz);
}
